// LongContextMoE_19043884990737
// MI455X (gfx1250) — compile-verified, matrix operand order checked by reading, not run on hardware
//
#include <hip/hip_runtime.h>

// ---------------------------------------------------------------------------
// Types / helpers
// ---------------------------------------------------------------------------
typedef unsigned short u16;
typedef unsigned long long u64;
typedef __attribute__((ext_vector_type(16))) __bf16 v16bf;
typedef __attribute__((ext_vector_type(8)))  float  v8f;

struct Pack32  { uint4 a, b; };          // 32 bytes
struct Pack16u { u16 s[16]; };           // 32 bytes

static __device__ __forceinline__ u16 f2bf(float x) {
  unsigned u = __builtin_bit_cast(unsigned, x);
  u += 0x7FFFu + ((u >> 16) & 1u);       // round-to-nearest-even
  return (u16)(u >> 16);
}

// load a v16bf A/B fragment: two contiguous 16-byte chunks (k0..k0+7, k0+16..k0+23)
static __device__ __forceinline__ v16bf frag16(const u16* p0, const u16* p1) {
  Pack32 u;
  u.a = *(const uint4*)p0;
  u.b = *(const uint4*)p1;
  return __builtin_bit_cast(v16bf, u);
}

static __device__ __forceinline__ v8f wmma_bf16(v16bf a, v16bf b, v8f c) {
  return __builtin_amdgcn_wmma_f32_16x16x32_bf16(false, a, false, b, (short)0, c,
                                                 false, false);
}

// --- CDNA5 async global->LDS copy (ASYNCcnt-tracked, no VGPR round-trip) ----
static __device__ __forceinline__ unsigned lds_addr32(const void* p) {
  // generic LDS pointer: low 32 bits = group-segment byte offset
  return (unsigned)(uintptr_t)p;
}
static __device__ __forceinline__ void async_b128(unsigned lds, const void* g) {
  u64 ga = (u64)(uintptr_t)g;
  asm volatile("global_load_async_to_lds_b128 %0, %1, off"
               :: "v"(lds), "v"(ga) : "memory");
}
static __device__ __forceinline__ void wait_async_le3() {
  asm volatile("s_wait_asynccnt 3" ::: "memory");
}
static __device__ __forceinline__ void wait_async_le8() {
  asm volatile("s_wait_asynccnt 8" ::: "memory");
}
static __device__ __forceinline__ void wait_async_0() {
  asm volatile("s_wait_asynccnt 0" ::: "memory");
}

// problem constants
constexpr int cB = 8, cS = 1024, cD = 512, cHH = 2048, cE = 4, cNH = 8, cDH = 64;
constexpr int cM = cB * cS;              // 8192 rows
constexpr int c3D = 3 * cD;              // 1536
constexpr int cSP = cS + 2;              // padded rows per batch (conv)

// ---------------------------------------------------------------------------
// Generic bf16 WMMA GEMM:  C[m,n] = sum_k A[m,k] * B[n,k]   (A:[M,K], B:[N,K])
// block tile 128(M) x 64(N), 8 waves of 32x32, K-step 32.
// Double-buffered LDS tiles filled with async global->LDS b128 copies:
// each thread issues exactly 3 async ops per K-step (2 for A, 1 for B).
// OUTMODE: 0 = store f32, 1 = store bf16 (padded-row mapping), 2 = outF += scale[b]*v
// ---------------------------------------------------------------------------
template <int OUTMODE, bool HAS_BIAS, bool HAS_CIN, bool RELU>
__global__ __launch_bounds__(256) void gemm_bf16_k(
    const u16* __restrict__ A, int lda, int aGap, int aOff,
    const u16* __restrict__ Bw, int ldb,
    const float* __restrict__ Cin, const float* __restrict__ bias,
    float* __restrict__ outF, u16* __restrict__ outH,
    int ldo, int oGap, int oOff,
    const float* __restrict__ scale, int rowsPerBatch, int K) {
  __shared__ u16 As[2][128 * 32];
  __shared__ u16 Bs[2][64 * 32];

  const int tid = threadIdx.x;
  const int lane = tid & 31;
  const int wid = tid >> 5;
  const int m0 = blockIdx.y * 128;
  const int n0 = blockIdx.x * 64;
  const int wm = (wid >> 1) * 32;
  const int wn = (wid & 1) * 32;

  v8f acc00 = {}, acc01 = {}, acc10 = {}, acc11 = {};

  // A tile staging: 128 rows x 32 cols, 2 threads/row, 2 x b128 each
  const int arow = tid >> 1;
  const int ac0 = (tid & 1) * 16;
  const int gm = m0 + arow;
  const long aPhys = (long)gm + (long)(gm / rowsPerBatch) * aGap + aOff;
  const u16* aRow = A + (size_t)aPhys * lda;
  // B tile staging: 64 rows x 32 cols, 4 threads/row, 1 x b128 each
  const int brow = tid >> 2;
  const int bc0 = (tid & 3) * 8;
  const u16* bRow = Bw + (size_t)(n0 + brow) * ldb;

  const int lr = lane & 15;
  const int lo_ = (lane & 16) ? 8 : 0;

  auto issue_tile = [&](int k0, int buf) {
    const u16* ga = aRow + k0 + ac0;
    async_b128(lds_addr32(&As[buf][arow * 32 + ac0]), ga);
    async_b128(lds_addr32(&As[buf][arow * 32 + ac0 + 8]), ga + 8);
    async_b128(lds_addr32(&Bs[buf][brow * 32 + bc0]), bRow + k0 + bc0);
  };

  issue_tile(0, 0);
  int buf = 0;
  for (int k0 = 0; k0 < K; k0 += 32) {
    const bool has_next = (k0 + 32 < K);
    if (has_next) {
      issue_tile(k0 + 32, buf ^ 1);
      wait_async_le3();        // previous tile's 3 copies complete (in-order)
    } else {
      wait_async_0();
    }
    __syncthreads();           // whole block's copies visible

    const u16* a0 = &As[buf][(wm + lr) * 32 + lo_];
    const u16* a1 = &As[buf][(wm + 16 + lr) * 32 + lo_];
    const u16* b0 = &Bs[buf][(wn + lr) * 32 + lo_];
    const u16* b1 = &Bs[buf][(wn + 16 + lr) * 32 + lo_];
    v16bf af0 = frag16(a0, a0 + 16);
    v16bf af1 = frag16(a1, a1 + 16);
    v16bf bf0 = frag16(b0, b0 + 16);
    v16bf bf1 = frag16(b1, b1 + 16);

    acc00 = wmma_bf16(af0, bf0, acc00);
    acc01 = wmma_bf16(af0, bf1, acc01);
    acc10 = wmma_bf16(af1, bf0, acc10);
    acc11 = wmma_bf16(af1, bf1, acc11);
    __syncthreads();           // protect buffer before next-next issue
    buf ^= 1;
  }

#pragma unroll
  for (int i = 0; i < 2; ++i)
#pragma unroll
    for (int j = 0; j < 2; ++j) {
      v8f& a = (i == 0) ? (j == 0 ? acc00 : acc01) : (j == 0 ? acc10 : acc11);
      const int n = n0 + wn + j * 16 + lr;
#pragma unroll
      for (int e = 0; e < 8; ++e) {
        const int m = m0 + wm + i * 16 + e + lo_;
        float v = a[e];
        if (HAS_CIN) v += Cin[(size_t)m * ldo + n];
        if (HAS_BIAS) v += bias[n];
        if (RELU) v = fmaxf(v, 0.f);
        if (OUTMODE == 0) {
          outF[(size_t)m * ldo + n] = v;
        } else if (OUTMODE == 1) {
          long p = (long)m + (long)(m / rowsPerBatch) * oGap + oOff;
          outH[(size_t)p * ldo + n] = f2bf(v);
        } else {
          const float sc = scale[(m / rowsPerBatch) * 16];
          outF[(size_t)m * ldo + n] += sc * v;
        }
      }
    }
}

// ---------------------------------------------------------------------------
// Flash-style windowed attention, 1 wave per 16-query tile.
// qkv: bf16 [B*S, 3D] interleaved (q|k|v, head h uses cols h*64..h*64+63)
// att: bf16 [B*S, D] output (1/sqrt(DH) scaled softmax(QK^T) V, head-concat)
// S^T tiles via wmma(K_frag, Q_frag) => P lands directly in A-frag layout.
// V tiles double-buffered in LDS via async b128 copies (8 per wave per tile).
// ---------------------------------------------------------------------------
__global__ __launch_bounds__(32) void attn_k(
    const u16* __restrict__ qkv, u16* __restrict__ att,
    int kv0, int kvLen, int q0, int outRow0, int qTiles) {
  const int lane = threadIdx.x;
  const int qt = blockIdx.x % qTiles;
  const int h = (blockIdx.x / qTiles) % cNH;
  const int b = blockIdx.x / (qTiles * cNH);

  const int lr = lane & 15;
  const int lo_ = (lane & 16) ? 8 : 0;
  const u16* base = qkv + (size_t)b * cS * c3D;

  // Q fragments (B-operand; lane = query row)
  const u16* qr = base + (size_t)(q0 + qt * 16 + lr) * c3D + h * cDH + lo_;
  const v16bf qf0 = frag16(qr, qr + 16);        // d = 0..31
  const v16bf qf1 = frag16(qr + 32, qr + 48);   // d = 32..63

  v8f o[4] = {};
  float run_m = -1e30f, run_l = 0.f;

  __shared__ u16 vs[2][32 * 64];

  auto issue_v = [&](int kc, int buf) {
    const u16* vsrc = base + (size_t)(kv0 + kc + lane) * c3D + 2 * cD + h * cDH;
#pragma unroll
    for (int i = 0; i < 8; ++i)
      async_b128(lds_addr32(&vs[buf][lane * 64 + i * 8]), vsrc + i * 8);
  };

  const int nt = kvLen / 32;
  issue_v(0, 0);
  int buf = 0;
  for (int t = 0; t < nt; ++t) {
    if (t + 1 < nt) {
      issue_v((t + 1) * 32, buf ^ 1);
      wait_async_le8();        // previous tile's 8 copies complete
    } else {
      wait_async_0();
    }
    const int kc = t * 32;

    // S^T tiles for k-sub 0 / 1
    v8f st[2];
#pragma unroll
    for (int tt = 0; tt < 2; ++tt) {
      const u16* kr =
          base + (size_t)(kv0 + kc + tt * 16 + lr) * c3D + cD + h * cDH + lo_;
      v16bf kf0 = frag16(kr, kr + 16);
      v16bf kf1 = frag16(kr + 32, kr + 48);
      v8f c = {};
      c = wmma_bf16(kf0, qf0, c);
      c = wmma_bf16(kf1, qf1, c);
      st[tt] = c;
    }

    // online softmax (lane & lane^16 share query row => single xor-16 reduce)
    float tmax = -1e30f;
#pragma unroll
    for (int tt = 0; tt < 2; ++tt)
#pragma unroll
      for (int e = 0; e < 8; ++e) {
        float s = st[tt][e] * 0.125f;  // 1/sqrt(64)
        st[tt][e] = s;
        tmax = fmaxf(tmax, s);
      }
    tmax = fmaxf(tmax, __shfl_xor(tmax, 16, 32));
    const float newm = fmaxf(run_m, tmax);
    const float f = __expf(run_m - newm);
    float lsum = 0.f;
#pragma unroll
    for (int tt = 0; tt < 2; ++tt)
#pragma unroll
      for (int e = 0; e < 8; ++e) {
        float p = __expf(st[tt][e] - newm);
        st[tt][e] = p;
        lsum += p;
      }
    lsum += __shfl_xor(lsum, 16, 32);
    run_l = run_l * f + lsum;
    run_m = newm;

    // rescale accumulators (factor for row q = e + lo_ lives at lane q)
#pragma unroll
    for (int e = 0; e < 8; ++e) {
      const float fe = __shfl(f, e + lo_, 32);
      o[0][e] *= fe; o[1][e] *= fe; o[2][e] *= fe; o[3][e] *= fe;
    }

    // P fragment: S^T register layout == A-fragment layout after f32->bf16
    Pack16u pu;
#pragma unroll
    for (int e = 0; e < 8; ++e) {
      pu.s[e] = f2bf(st[0][e]);
      pu.s[8 + e] = f2bf(st[1][e]);
    }
    const v16bf pf = __builtin_bit_cast(v16bf, pu);

    // O += P @ V  (V^T fragments gathered from LDS; lane = output d column)
#pragma unroll
    for (int n = 0; n < 4; ++n) {
      Pack16u vu;
      const int d = n * 16 + lr;
#pragma unroll
      for (int e = 0; e < 8; ++e) {
        vu.s[e] = vs[buf][(lo_ + e) * 64 + d];
        vu.s[8 + e] = vs[buf][(16 + lo_ + e) * 64 + d];
      }
      const v16bf vf = __builtin_bit_cast(v16bf, vu);
      o[n] = wmma_bf16(pf, vf, o[n]);
    }
    buf ^= 1;
  }

  const float il = 1.f / run_l;
#pragma unroll
  for (int e = 0; e < 8; ++e) {
    const float ie = __shfl(il, e + lo_, 32);
    const int row = outRow0 + qt * 16 + e + lo_;
#pragma unroll
    for (int n = 0; n < 4; ++n)
      att[((size_t)b * cS + row) * cD + h * cDH + n * 16 + lr] =
          f2bf(o[n][e] * ie);
  }
}

// ---------------------------------------------------------------------------
// Small elementwise / router kernels
// ---------------------------------------------------------------------------
__global__ void zero_f32_k(float* p, long n) {
  long i = (long)blockIdx.x * blockDim.x + threadIdx.x;
  if (i < n) p[i] = 0.f;
}
__global__ void zero_u16_k(u16* p, long n) {
  long i = (long)blockIdx.x * blockDim.x + threadIdx.x;
  if (i < n) p[i] = 0;
}
__global__ void cvt_k(const float* __restrict__ s, u16* __restrict__ d, long n) {
  long i = (long)blockIdx.x * blockDim.x + threadIdx.x;
  if (i < n) d[i] = f2bf(s[i]);
}
// x [B,S,D] -> zero-padded bf16 [B, S+2, D]
__global__ void padx_k(const float* __restrict__ x, u16* __restrict__ xp) {
  long i = (long)blockIdx.x * blockDim.x + threadIdx.x;
  const long n = (long)cB * cSP * cD;
  if (i >= n) return;
  const int d = (int)(i % cD);
  const long r = i / cD;
  const int sp = (int)(r % cSP);
  const int b = (int)(r / cSP);
  float v = 0.f;
  if (sp >= 1 && sp <= cS) v = x[((size_t)b * cS + (sp - 1)) * cD + d];
  xp[i] = f2bf(v);
}
// conv weight [O, I, 3] -> per-tap packed bf16 [3][O][I]
__global__ void repack3_k(const float* __restrict__ src, u16* __restrict__ dst,
                          int O, int I) {
  long i = (long)blockIdx.x * blockDim.x + threadIdx.x;
  const long OI = (long)O * I;
  if (i >= 3 * OI) return;
  const int t = (int)(i / OI);
  const long rem = i - (long)t * OI;
  const long o = rem / I;
  const long ii = rem - o * I;
  dst[i] = f2bf(src[(o * I + ii) * 3 + t]);
}
__global__ void mean_k(const float* __restrict__ x, float* __restrict__ xm) {
  int i = blockIdx.x * blockDim.x + threadIdx.x;
  if (i >= cB * cD) return;
  const int b = i >> 9, d = i & (cD - 1);
  const float* p = x + (size_t)b * cS * cD + d;
  float s = 0.f;
  for (int t = 0; t < cS; ++t) s += p[(size_t)t * cD];
  xm[i] = s * (1.f / (float)cS);
}
__global__ void router_h_k(const float* __restrict__ xm,
                           const float* __restrict__ w1,
                           const float* __restrict__ b1,
                           float* __restrict__ rh) {
  int i = blockIdx.x * blockDim.x + threadIdx.x;
  if (i >= cB * cHH) return;
  const int b = i / cHH, o = i % cHH;
  const float* xv = xm + (size_t)b * cD;
  const float* wv = w1 + (size_t)o * cD;
  float s = b1[o];
  for (int d = 0; d < cD; ++d) s += xv[d] * wv[d];
  rh[i] = fmaxf(s, 0.f);
}
__global__ void router_sm_k(const float* __restrict__ rh,
                            const float* __restrict__ w2,
                            const float* __restrict__ b2,
                            float* __restrict__ probs) {
  __shared__ float lg[16];
  const int b = blockIdx.x, t = threadIdx.x;
  if (t < 16) {
    const float* r = rh + (size_t)b * cHH;
    const float* w = w2 + (size_t)t * cHH;
    float s = b2[t];
    for (int o = 0; o < cHH; ++o) s += r[o] * w[o];
    lg[t] = s;
  }
  __syncthreads();
  if (t == 0) {
    float mx = lg[0];
    for (int k = 1; k < 16; ++k) mx = fmaxf(mx, lg[k]);
    float e[16], sum = 0.f;
    for (int k = 0; k < 16; ++k) { e[k] = __expf(lg[k] - mx); sum += e[k]; }
    const float inv = 1.f / sum;
    for (int k = 0; k < 16; ++k) probs[b * 16 + k] = e[k] * inv;
  }
}

// ---------------------------------------------------------------------------
// Host orchestration
// ---------------------------------------------------------------------------
static inline int eb(long n) { return (int)((n + 255) / 256); }

extern "C" void kernel_launch(void* const* d_in, const int* in_sizes, int n_in,
                              void* d_out, int out_size, void* d_ws, size_t ws_size,
                              hipStream_t stream) {
  (void)in_sizes; (void)n_in; (void)ws_size;
  const float* x       = (const float*)d_in[0];
  const float* rw1     = (const float*)d_in[1];
  const float* rb1     = (const float*)d_in[2];
  const float* rw2     = (const float*)d_in[3];
  const float* rb2     = (const float*)d_in[4];
  const float* mlp_w1  = (const float*)d_in[5];
  const float* mlp_b1  = (const float*)d_in[6];
  const float* mlp_w2  = (const float*)d_in[7];
  const float* mlp_b2  = (const float*)d_in[8];
  const float* conv_w1 = (const float*)d_in[9];
  const float* conv_b1 = (const float*)d_in[10];
  const float* conv_w2 = (const float*)d_in[11];
  const float* conv_b2 = (const float*)d_in[12];
  const float* tr_wqkv = (const float*)d_in[13];
  const float* tr_bqkv = (const float*)d_in[14];
  const float* tr_wout = (const float*)d_in[15];
  const float* tr_bout = (const float*)d_in[16];
  const float* hi_wqkv = (const float*)d_in[17];
  const float* hi_bqkv = (const float*)d_in[18];
  const float* hi_wout = (const float*)d_in[19];
  const float* hi_bout = (const float*)d_in[20];
  float* out = (float*)d_out;

  // workspace layout
  char* w = (char*)d_ws;
  size_t off = 0;
  auto alloc = [&](size_t bytes) { size_t o = off; off = (off + bytes + 255) & ~(size_t)255; return o; };
  u16*   Xb    = (u16*)(w + alloc((size_t)cM * cD * 2));
  u16*   Hb    = (u16*)(w + alloc((size_t)cM * cHH * 2));
  u16*   Xpad  = (u16*)(w + alloc((size_t)cB * cSP * cD * 2));
  u16*   Hpad  = (u16*)(w + alloc((size_t)cB * cSP * cHH * 2));
  float* Tf32  = (float*)(w + alloc((size_t)cM * cHH * 4));
  u16*   qkvb  = (u16*)(w + alloc((size_t)cM * c3D * 2));
  u16*   attb  = (u16*)(w + alloc((size_t)cM * cD * 2));
  u16*   curb  = (u16*)(w + alloc((size_t)cM * cD * 2));
  u16*   wbuf  = (u16*)(w + alloc((size_t)3 * cHH * cD * 2));
  float* xm    = (float*)(w + alloc((size_t)cB * cD * 4));
  float* rh    = (float*)(w + alloc((size_t)cB * cHH * 4));
  float* probs = (float*)(w + alloc((size_t)cB * 16 * 4));

  const dim3 blk(256);
  const dim3 gH(cHH / 64, cM / 128);   // N=2048
  const dim3 gD(cD / 64, cM / 128);    // N=512
  const dim3 gQ(c3D / 64, cM / 128);   // N=1536

  // ---- prep ---------------------------------------------------------------
  zero_f32_k<<<eb(out_size), blk, 0, stream>>>(out, out_size);
  mean_k<<<eb(cB * cD), blk, 0, stream>>>(x, xm);
  router_h_k<<<eb(cB * cHH), blk, 0, stream>>>(xm, rw1, rb1, rh);
  router_sm_k<<<cB, 32, 0, stream>>>(rh, rw2, rb2, probs);
  cvt_k<<<eb((long)cM * cD), blk, 0, stream>>>(x, Xb, (long)cM * cD);
  padx_k<<<eb((long)cB * cSP * cD), blk, 0, stream>>>(x, Xpad);
  zero_u16_k<<<eb((long)cB * cSP * cHH), blk, 0, stream>>>(Hpad, (long)cB * cSP * cHH);

  // ---- type 0: MLP experts ------------------------------------------------
  for (int e = 0; e < cE; ++e) {
    cvt_k<<<eb((long)cHH * cD), blk, 0, stream>>>(mlp_w1 + (size_t)e * cHH * cD, wbuf, (long)cHH * cD);
    gemm_bf16_k<1, true, false, true><<<gH, blk, 0, stream>>>(
        Xb, cD, 0, 0, wbuf, cD, nullptr, mlp_b1 + e * cHH,
        nullptr, Hb, cHH, 0, 0, nullptr, cS, cD);
    cvt_k<<<eb((long)cD * cHH), blk, 0, stream>>>(mlp_w2 + (size_t)e * cD * cHH, wbuf, (long)cD * cHH);
    gemm_bf16_k<2, true, false, false><<<gD, blk, 0, stream>>>(
        Hb, cHH, 0, 0, wbuf, cHH, nullptr, mlp_b2 + e * cD,
        out, nullptr, cD, 0, 0, probs + 0 * cE + e, cS, cHH);
  }

  // ---- type 1: conv experts (k=3 as 3 shifted GEMM taps) -------------------
  for (int e = 0; e < cE; ++e) {
    repack3_k<<<eb((long)3 * cHH * cD), blk, 0, stream>>>(
        conv_w1 + (size_t)e * cHH * cD * 3, wbuf, cHH, cD);
    gemm_bf16_k<0, false, false, false><<<gH, blk, 0, stream>>>(
        Xpad, cD, 2, 0, wbuf + 0 * (size_t)cHH * cD, cD, nullptr, nullptr,
        Tf32, nullptr, cHH, 0, 0, nullptr, cS, cD);
    gemm_bf16_k<0, false, true, false><<<gH, blk, 0, stream>>>(
        Xpad, cD, 2, 1, wbuf + 1 * (size_t)cHH * cD, cD, Tf32, nullptr,
        Tf32, nullptr, cHH, 0, 0, nullptr, cS, cD);
    gemm_bf16_k<1, true, true, true><<<gH, blk, 0, stream>>>(
        Xpad, cD, 2, 2, wbuf + 2 * (size_t)cHH * cD, cD, Tf32, conv_b1 + e * cHH,
        nullptr, Hpad, cHH, 2, 1, nullptr, cS, cD);

    repack3_k<<<eb((long)3 * cD * cHH), blk, 0, stream>>>(
        conv_w2 + (size_t)e * cD * cHH * 3, wbuf, cD, cHH);
    gemm_bf16_k<0, false, false, false><<<gD, blk, 0, stream>>>(
        Hpad, cHH, 2, 0, wbuf + 0 * (size_t)cD * cHH, cHH, nullptr, nullptr,
        Tf32, nullptr, cD, 0, 0, nullptr, cS, cHH);
    gemm_bf16_k<0, false, true, false><<<gD, blk, 0, stream>>>(
        Hpad, cHH, 2, 1, wbuf + 1 * (size_t)cD * cHH, cHH, Tf32, nullptr,
        Tf32, nullptr, cD, 0, 0, nullptr, cS, cHH);
    gemm_bf16_k<2, true, true, false><<<gD, blk, 0, stream>>>(
        Hpad, cHH, 2, 2, wbuf + 2 * (size_t)cD * cHH, cHH, Tf32, conv_b2 + e * cD,
        out, nullptr, cD, 0, 0, probs + 1 * cE + e, cS, cHH);
  }

  // ---- type 2: transformer experts (full MHA) ------------------------------
  for (int e = 0; e < cE; ++e) {
    cvt_k<<<eb((long)c3D * cD), blk, 0, stream>>>(tr_wqkv + (size_t)e * c3D * cD, wbuf, (long)c3D * cD);
    gemm_bf16_k<1, true, false, false><<<gQ, blk, 0, stream>>>(
        Xb, cD, 0, 0, wbuf, cD, nullptr, tr_bqkv + e * c3D,
        nullptr, qkvb, c3D, 0, 0, nullptr, cS, cD);
    attn_k<<<cB * cNH * 64, 32, 0, stream>>>(qkvb, attb, 0, cS, 0, 0, 64);
    cvt_k<<<eb((long)cD * cD), blk, 0, stream>>>(tr_wout + (size_t)e * cD * cD, wbuf, (long)cD * cD);
    gemm_bf16_k<2, true, false, false><<<gD, blk, 0, stream>>>(
        attb, cD, 0, 0, wbuf, cD, nullptr, tr_bout + e * cD,
        out, nullptr, cD, 0, 0, probs + 2 * cE + e, cS, cD);
  }

  // ---- type 3: hierarchical experts ---------------------------------------
  // At S=1024: level 0 = two windows ([0,512) and q/kv [448,960) -> rows 512..1023),
  // levels 1 & 2 = full MHA (later chunks are truncated away by [:, :S]).
  for (int e = 0; e < cE; ++e) {
    const u16* cur = Xb;
    for (int l = 0; l < 3; ++l) {
      const size_t wi = (size_t)(e * 3 + l);
      cvt_k<<<eb((long)c3D * cD), blk, 0, stream>>>(hi_wqkv + wi * c3D * cD, wbuf, (long)c3D * cD);
      gemm_bf16_k<1, true, false, false><<<gQ, blk, 0, stream>>>(
          cur, cD, 0, 0, wbuf, cD, nullptr, hi_bqkv + wi * c3D,
          nullptr, qkvb, c3D, 0, 0, nullptr, cS, cD);
      if (l == 0) {
        attn_k<<<cB * cNH * 32, 32, 0, stream>>>(qkvb, attb, 0, 512, 0, 0, 32);
        attn_k<<<cB * cNH * 32, 32, 0, stream>>>(qkvb, attb, 448, 512, 448, 512, 32);
      } else {
        attn_k<<<cB * cNH * 64, 32, 0, stream>>>(qkvb, attb, 0, cS, 0, 0, 64);
      }
      cvt_k<<<eb((long)cD * cD), blk, 0, stream>>>(hi_wout + wi * cD * cD, wbuf, (long)cD * cD);
      if (l < 2) {
        gemm_bf16_k<1, true, false, false><<<gD, blk, 0, stream>>>(
            attb, cD, 0, 0, wbuf, cD, nullptr, hi_bout + wi * cD,
            nullptr, curb, cD, 0, 0, nullptr, cS, cD);
        cur = curb;
      } else {
        gemm_bf16_k<2, true, false, false><<<gD, blk, 0, stream>>>(
            attb, cD, 0, 0, wbuf, cD, nullptr, hi_bout + wi * cD,
            out, nullptr, cD, 0, 0, probs + 3 * cE + e, cS, cD);
      }
    }
  }
}
